// TileMatching_76304388980752
// MI455X (gfx1250) — compile-verified, matrix operand order checked by reading, not run on hardware
//
#include <hip/hip_runtime.h>

typedef __attribute__((ext_vector_type(16))) _Float16 v16h;
typedef __attribute__((ext_vector_type(8)))  float    v8f;

// Problem constants from the reference setup
constexpr int Bn   = 4;
constexpr int Cc   = 3;
constexpr int Hh   = 512;
constexpr int Ww   = 512;
constexpr int T    = 16;   // TILE
constexpr int Rr   = 8;    // search radius
constexpr int ND   = 17;   // 2R+1 displacements per axis
constexpr int NT   = 32;   // tiles per axis (512/16)
constexpr int WIN  = 32;   // window = T + 2R
constexpr int WROW = 36;   // padded LDS row stride (36*r mod 64 distinct for r=0..15)
constexpr int CHW  = WIN * WROW; // floats per channel in LDS window

// |x| via an opaque VALU max-with-neg-modifier. Inline asm is deliberate:
// every source-level fabs idiom (fabsf, fmaxf(x,-x), bit masking) gets
// canonicalized to llvm.fabs, SLP-vectorized to v2f32, and crashes this
// gfx1250 backend's instruction selector on the sign-mask bitcast/and node.
__device__ __forceinline__ float abs_opaque(float x) {
    float r;
    asm("v_max_num_f32 %0, %1, -%1" : "=v"(r) : "v"(x));
    return r;
}

__global__ __launch_bounds__(256)
void tile_match_wmma(const float* __restrict__ src,
                     const float* __restrict__ dst,
                     const int*   __restrict__ offset,
                     float* __restrict__ outOff,
                     float* __restrict__ outAligned)
{
    __shared__ float win[Cc * CHW];     // 3 x 32x36 zero-padded dst window (~13.8 KB)
    __shared__ float redC[8];
    __shared__ int   redD[8];
    __shared__ int   s_best;

    const int tx  = blockIdx.x;
    const int ty  = blockIdx.y;
    const int b   = blockIdx.z;
    const int tid = threadIdx.x;
    const int wv  = tid >> 5;           // wave id (wave32)
    const int lane = tid & 31;
    const int ti = ty * T, tj = tx * T;

    const int offY = offset[((b * 2 + 0) * NT + ty) * NT + tx];
    const int offX = offset[((b * 2 + 1) * NT + ty) * NT + tx];

    // ---- Stage dst search window into LDS with zero padding (coalesced) ----
    for (int idx = tid; idx < Cc * WIN * WIN; idx += 256) {
        int c   = idx / (WIN * WIN);
        int rem = idx - c * (WIN * WIN);
        int wy  = rem >> 5, wx = rem & 31;
        int gr  = ti + offY - Rr + wy;
        int gc  = tj + offX - Rr + wx;
        float v = 0.0f;
        if ((unsigned)gr < (unsigned)Hh && (unsigned)gc < (unsigned)Ww)
            v = dst[((b * Cc + c) * Hh + gr) * Ww + gc];
        win[c * CHW + wy * WROW + wx] = v;
    }

    // ---- Per-lane src tile preload (matches f16 WMMA A layout) ----
    // lanes 0-15: row m=lane, cols n=0..7 ; lanes 16-31: row m=lane-16, cols n=8..15
    const int half = lane >> 4;
    const int m    = lane & 15;
    const int n0   = half * 8;
    float s0[8], s1[8], s2[8];
    {
        const float* p0 = &src[((b * Cc + 0) * Hh + ti + m) * Ww + tj + n0];
        const float* p1 = &src[((b * Cc + 1) * Hh + ti + m) * Ww + tj + n0];
        const float* p2 = &src[((b * Cc + 2) * Hh + ti + m) * Ww + tj + n0];
        #pragma unroll
        for (int j = 0; j < 8; ++j) { s0[j] = p0[j]; s1[j] = p1[j]; s2[j] = p2[j]; }
    }

    __syncthreads();

    v16h ones;
    #pragma unroll
    for (int j = 0; j < 16; ++j) ones[j] = (_Float16)1.0f;

    const float* w0 = &win[0 * CHW + m * WROW + n0];
    const float* w1 = &win[1 * CHW + m * WROW + n0];
    const float* w2 = &win[2 * CHW + m * WROW + n0];

    float bestC = 3.4e38f;
    int   bestD = 0;

    // ---- Displacement search: wave w handles d = w, w+8, ... (ascending => tie-break ok)
    // Per displacement: VALU computes |cand-src| (f16), matrix pipe accumulates:
    //   D = A(16x32 f16, ch0|ch1 in K) * ones + [A(ch2|0) * ones]  with f32 C accumulation.
    for (int d = wv; d < ND * ND; d += 8) {
        int dy = d / ND, dx = d % ND;
        int doff = dy * WROW + dx;

        v16h a01, a2;
        #pragma unroll
        for (int j = 0; j < 8; ++j) {
            float d0 = w0[doff + j] - s0[j];
            float d1 = w1[doff + j] - s1[j];
            float d2 = w2[doff + j] - s2[j];
            a01[j]     = (_Float16)abs_opaque(d0);
            a01[8 + j] = (_Float16)abs_opaque(d1);
            a2[j]      = (_Float16)abs_opaque(d2);
            a2[8 + j]  = (_Float16)0.0f;
        }

        v8f acc = {};
        acc = __builtin_amdgcn_wmma_f32_16x16x32_f16(false, a01, false, ones,
                                                     (short)0, acc, false, false);
        acc = __builtin_amdgcn_wmma_f32_16x16x32_f16(false, a2, false, ones,
                                                     (short)0, acc, false, false);

        // acc[i] = full row-sum of rows in this lane's half; sum 8, add partner half
        float p = acc[0] + acc[1] + acc[2] + acc[3] + acc[4] + acc[5] + acc[6] + acc[7];
        float cost = p + __shfl_xor(p, 16, 32);

        if (cost < bestC) { bestC = cost; bestD = d; }
    }

    if (lane == 0) { redC[wv] = bestC; redD[wv] = bestD; }
    __syncthreads();

    if (tid == 0) {
        float bc = redC[0]; int bd = redD[0];
        #pragma unroll
        for (int i = 1; i < 8; ++i) {
            float c = redC[i]; int dd = redD[i];
            if (c < bc || (c == bc && dd < bd)) { bc = c; bd = dd; }
        }
        s_best = bd;
    }
    __syncthreads();

    const int bd  = s_best;
    const int bdy = bd / ND, bdx = bd % ND;

    // ---- Outputs ----
    if (tid == 0) {
        outOff[((b * 2 + 0) * NT + ty) * NT + tx] = (float)(offY + bdy - Rr);
        outOff[((b * 2 + 1) * NT + ty) * NT + tx] = (float)(offX + bdx - Rr);
    }
    {
        int mm = tid >> 4, nn = tid & 15;
        #pragma unroll
        for (int c = 0; c < Cc; ++c) {
            outAligned[((b * Cc + c) * Hh + ti + mm) * Ww + tj + nn] =
                win[c * CHW + (mm + bdy) * WROW + (nn + bdx)];
        }
    }
}

extern "C" void kernel_launch(void* const* d_in, const int* in_sizes, int n_in,
                              void* d_out, int out_size, void* d_ws, size_t ws_size,
                              hipStream_t stream) {
    const float* src    = (const float*)d_in[0];
    const float* dst    = (const float*)d_in[1];
    const int*   offset = (const int*)d_in[2];

    float* outOff     = (float*)d_out;                 // new_off: 4*2*32*32 = 8192
    float* outAligned = outOff + Bn * 2 * NT * NT;     // aligned: 4*3*512*512

    dim3 grid(NT, NT, Bn);   // one workgroup (8 wave32) per tile
    tile_match_wmma<<<grid, 256, 0, stream>>>(src, dst, offset, outOff, outAligned);
}
